// TransformerBlock_49160195670192
// MI455X (gfx1250) — matrix operand order repaired and checked by reading, not compiled, not run on hardware
//
#include <hip/hip_runtime.h>

// ---------------- problem constants ----------------
constexpr int MD = 512;          // model dim
constexpr int NH = 8;            // heads
constexpr int HD = 64;           // head dim
constexpr int SQ = 2048;         // sequence length
constexpr int NB = 4;            // batch
constexpr int NT = NB * SQ;      // tokens = 8192
constexpr int FF = 2048;         // mlp hidden = 4*MD

typedef __attribute__((ext_vector_type(16))) _Float16 v16h;
typedef __attribute__((ext_vector_type(8)))  _Float16 v8h;
typedef __attribute__((ext_vector_type(4)))  _Float16 v4h;
typedef __attribute__((ext_vector_type(8)))  float    v8f;

// A-matrix fragment (16x32 f16): lane m = lane&15, hsel = lane>>4.
// VGPR0-7 hold K = base..base+15,
// where base = k0 + hsel*16.  -> two 16B loads at p and p+8.
static __device__ __forceinline__ v16h load_fragA(const _Float16* p) {
  const v8h* q = (const v8h*)p;
  v8h lo = q[0];
  v8h hi = q[1];
  return __builtin_shufflevector(lo, hi, 0, 1, 2, 3, 4, 5, 6, 7,
                                         8, 9, 10, 11, 12, 13, 14, 15);
}

// B-matrix fragment (32x16 f16): lane n = lane&15; lanes 0-15 hold K=k0..k0+15,
// lanes 16-31 hold K=k0+16..k0+31 (contiguous 32B at p = row + k0 + hsel*16).
static __device__ __forceinline__ v16h load_fragB(const _Float16* p) {
  const v8h* q = (const v8h*)p;
  v8h lo = q[0];
  v8h hi = q[1];
  return __builtin_shufflevector(lo, hi, 0, 1, 2, 3, 4, 5, 6, 7,
                                         8, 9, 10, 11, 12, 13, 14, 15);
}

static __device__ __forceinline__ v8f wmma16(v16h a, v16h b, v8f c) {
  return __builtin_amdgcn_wmma_f32_16x16x32_f16(false, a, false, b,
                                                (short)0, c, false, false);
}

// ---------------- f32 -> f16 convert (vectorized x4) ----------------
__global__ void f32_to_f16_k(const float* __restrict__ in,
                             _Float16* __restrict__ out, int n4) {
  int i = blockIdx.x * blockDim.x + threadIdx.x;
  if (i < n4) {
    const float4 f = ((const float4*)in)[i];
    v4h o = {(_Float16)f.x, (_Float16)f.y, (_Float16)f.z, (_Float16)f.w};
    ((v4h*)out)[i] = o;
  }
}

// ------- V transpose via padded LDS tile: [tok, h*64+dh] -> [b,h][dh][s] -------
__global__ __launch_bounds__(256) void transpose_v_k(
    const _Float16* __restrict__ v, _Float16* __restrict__ vt) {
  __shared__ _Float16 tile[64][72];              // pad to dodge bank conflicts
  int blk = blockIdx.x;                          // NB*NH*(SQ/64) blocks
  const int st = blk & 31; blk >>= 5;            // s-tile index (SQ/64 = 32)
  const int h = blk & (NH - 1);
  const int b = blk >> 3;
  const int s0 = st * 64;
  const int t = threadIdx.x;
#pragma unroll
  for (int i = 0; i < 16; ++i) {                 // coalesced read along dh
    const int lin = t + i * 256;
    const int sl = lin >> 6, dl = lin & 63;
    tile[sl][dl] = v[(size_t)(b * SQ + s0 + sl) * MD + h * HD + dl];
  }
  __syncthreads();
#pragma unroll
  for (int i = 0; i < 16; ++i) {                 // coalesced write along s
    const int lin = t + i * 256;
    const int dl = lin >> 6, sl = lin & 63;
    vt[((size_t)(b * NH + h) * HD + dl) * SQ + s0 + sl] = tile[sl][dl];
  }
}

// ---------------- tiled GEMM: out[M,N] = A[M,K] @ W[N,K]^T ----------------
// block = 128 threads (4 waves); wave w owns rows [by*128 + 32w, +32), cols
// [bx*64, +64): 8 independent 16x16 WMMA accumulator chains (hazard cover).
template <bool BIAS, bool RELU, bool RES, bool OUTF, bool OUTH>
__global__ __launch_bounds__(128) void gemm_xwT(
    const _Float16* __restrict__ A, const _Float16* __restrict__ Bw,
    const float* __restrict__ bias, const float* __restrict__ res,
    float* __restrict__ outF, _Float16* __restrict__ outH,
    int M, int N, int K) {
  const int lane = threadIdx.x & 31;
  const int wave = threadIdx.x >> 5;
  const int hsel = lane >> 4;
  const int l15  = lane & 15;
  const int rowBase = blockIdx.y * 128 + wave * 32;
  const int colBase = blockIdx.x * 64;

  const _Float16* arow0 = A  + (size_t)(rowBase +  0 + l15) * K;
  const _Float16* arow1 = A  + (size_t)(rowBase + 16 + l15) * K;
  const _Float16* b0r   = Bw + (size_t)(colBase +  0 + l15) * K;
  const _Float16* b1r   = Bw + (size_t)(colBase + 16 + l15) * K;
  const _Float16* b2r   = Bw + (size_t)(colBase + 32 + l15) * K;
  const _Float16* b3r   = Bw + (size_t)(colBase + 48 + l15) * K;

  v8f c[8] = {};
  for (int k0 = 0; k0 < K; k0 += 32) {
    const int ka = k0 + hsel * 16;   // A packing
    const int kb = k0 + hsel * 16;   // B packing
    v16h a0 = load_fragA(arow0 + ka);
    v16h a1 = load_fragA(arow1 + ka);
    v16h b0 = load_fragB(b0r + kb);
    v16h b1 = load_fragB(b1r + kb);
    v16h b2 = load_fragB(b2r + kb);
    v16h b3 = load_fragB(b3r + kb);
    c[0] = wmma16(a0, b0, c[0]);
    c[4] = wmma16(a1, b0, c[4]);
    c[1] = wmma16(a0, b1, c[1]);
    c[5] = wmma16(a1, b1, c[5]);
    c[2] = wmma16(a0, b2, c[2]);
    c[6] = wmma16(a1, b2, c[6]);
    c[3] = wmma16(a0, b3, c[3]);
    c[7] = wmma16(a1, b3, c[7]);
  }

#pragma unroll
  for (int mt = 0; mt < 2; ++mt) {
#pragma unroll
    for (int sub = 0; sub < 4; ++sub) {
      v8f cc = c[mt * 4 + sub];
      const int n0 = colBase + sub * 16 + l15;
#pragma unroll
      for (int r = 0; r < 8; ++r) {
        const int m = rowBase + mt * 16 + hsel * 8 + r;  // C/D: M = r + 8*hsel
        float v = cc[r];
        if (BIAS) v += bias[n0];
        if (RELU) v = v > 0.f ? v : 0.f;
        if (RES)  v += res[(size_t)m * N + n0];
        if (OUTF) outF[(size_t)m * N + n0] = v;
        if (OUTH) outH[(size_t)m * N + n0] = (_Float16)v;
      }
    }
  }
}

// ---------------- attention: one block per (b, h, 16-row tile) ----------------
__global__ __launch_bounds__(128) void attention_k(
    const _Float16* __restrict__ q16, const _Float16* __restrict__ k16,
    const _Float16* __restrict__ vt, _Float16* __restrict__ ctx16) {
  extern __shared__ float sc[];                 // 16 x SQ f32 = 128 KB
  const int lane = threadIdx.x & 31;
  const int wave = threadIdx.x >> 5;
  const int hsel = lane >> 4;
  const int l15  = lane & 15;

  int blk = blockIdx.x;
  const int rt = blk & (SQ / 16 - 1);           // row tile (0..127)
  blk >>= 7;
  const int h = blk & (NH - 1);
  const int b = blk >> 3;
  const size_t tokBase = (size_t)b * SQ;

  // Q fragments for this wave (shared rows of the tile), K split 0..31 / 32..63
  const _Float16* qrow = q16 + (tokBase + rt * 16 + l15) * MD + h * HD;
  v16h qf0 = load_fragA(qrow + 0  + hsel * 16);
  v16h qf1 = load_fragA(qrow + 32 + hsel * 16);

  // ---- phase 1: scores = Q @ K^T into LDS (2 column tiles per iteration) ----
  for (int it = 0; it < SQ / 128; ++it) {       // 16 iterations, 8 tiles each
    const int ct0 = it * 8 + wave * 2;
    const int ct1 = ct0 + 1;
    const _Float16* kr0 = k16 + (tokBase + ct0 * 16 + l15) * MD + h * HD;
    const _Float16* kr1 = k16 + (tokBase + ct1 * 16 + l15) * MD + h * HD;
    v16h ka0 = load_fragB(kr0 + 0  + hsel * 16);
    v16h ka1 = load_fragB(kr0 + 32 + hsel * 16);
    v16h kb0 = load_fragB(kr1 + 0  + hsel * 16);
    v16h kb1 = load_fragB(kr1 + 32 + hsel * 16);
    v8f s0 = {}, s1 = {};
    s0 = wmma16(qf0, ka0, s0);
    s1 = wmma16(qf0, kb0, s1);
    s0 = wmma16(qf1, ka1, s0);
    s1 = wmma16(qf1, kb1, s1);
#pragma unroll
    for (int r = 0; r < 8; ++r) {
      sc[(hsel * 8 + r) * SQ + ct0 * 16 + l15] = s0[r];
      sc[(hsel * 8 + r) * SQ + ct1 * 16 + l15] = s1[r];
    }
  }
  __syncthreads();

  // ---- phase 2: softmax rows (8 lanes per row, within one wave) ----
  {
    const int row = threadIdx.x >> 3;
    const int sub = threadIdx.x & 7;
    float* srow = sc + row * SQ;
    const float scale = 0.125f;                 // 1/sqrt(64)
    float mx = -3.4e38f;
    for (int c = sub; c < SQ; c += 8) mx = fmaxf(mx, srow[c] * scale);
    mx = fmaxf(mx, __shfl_xor(mx, 1, 32));
    mx = fmaxf(mx, __shfl_xor(mx, 2, 32));
    mx = fmaxf(mx, __shfl_xor(mx, 4, 32));
    float sum = 0.f;
    for (int c = sub; c < SQ; c += 8) {
      float e = __expf(srow[c] * scale - mx);
      srow[c] = e;
      sum += e;
    }
    sum += __shfl_xor(sum, 1, 32);
    sum += __shfl_xor(sum, 2, 32);
    sum += __shfl_xor(sum, 4, 32);
    const float inv = 1.f / sum;
    for (int c = sub; c < SQ; c += 8) srow[c] *= inv;
  }
  __syncthreads();

  // ---- phase 3: ctx = P @ V; wave w owns DH columns [16w, 16w+16) ----
  const _Float16* vrow =
      vt + ((size_t)(b * NH + h) * HD + wave * 16 + l15) * SQ;
  const float* prow = sc + l15 * SQ;
  v8f acc0 = {}, acc1 = {};
  for (int k0 = 0; k0 < SQ; k0 += 64) {         // two interleaved chains
    const float* pp0 = prow + k0 + hsel * 16;
    const float* pp1 = pp0 + 32;
    v16h pa0, pa1;
#pragma unroll
    for (int i = 0; i < 8; ++i) {
      pa0[i]     = (_Float16)pp0[i];
      pa0[i + 8] = (_Float16)pp0[i + 8];
      pa1[i]     = (_Float16)pp1[i];
      pa1[i + 8] = (_Float16)pp1[i + 8];
    }
    v16h vb0 = load_fragB(vrow + k0 + hsel * 16);
    v16h vb1 = load_fragB(vrow + k0 + 32 + hsel * 16);
    acc0 = wmma16(pa0, vb0, acc0);
    acc1 = wmma16(pa1, vb1, acc1);
  }
  v8f acc = acc0 + acc1;
#pragma unroll
  for (int r = 0; r < 8; ++r) {
    const size_t tok = tokBase + rt * 16 + hsel * 8 + r;
    ctx16[tok * MD + h * HD + wave * 16 + l15] = (_Float16)acc[r];
  }
}

// ------------- layernorm over last dim (512), one token per block -------------
__global__ __launch_bounds__(128) void layernorm_k(
    const float* __restrict__ in, const float* __restrict__ g,
    const float* __restrict__ bb, float* __restrict__ outF,
    _Float16* __restrict__ outH) {
  __shared__ float red0[4], red1[4];
  const int tok = blockIdx.x;
  const int t = threadIdx.x;
  const float* row = in + (size_t)tok * MD;
  float v[4];
  float s = 0.f;
#pragma unroll
  for (int i = 0; i < 4; ++i) { v[i] = row[t + i * 128]; s += v[i]; }
#pragma unroll
  for (int m = 16; m >= 1; m >>= 1) s += __shfl_xor(s, m, 32);
  if ((t & 31) == 0) red0[t >> 5] = s;
  __syncthreads();
  const float mean = (red0[0] + red0[1] + red0[2] + red0[3]) * (1.f / MD);
  float vs = 0.f;
#pragma unroll
  for (int i = 0; i < 4; ++i) { float d = v[i] - mean; vs += d * d; }
#pragma unroll
  for (int m = 16; m >= 1; m >>= 1) vs += __shfl_xor(vs, m, 32);
  if ((t & 31) == 0) red1[t >> 5] = vs;
  __syncthreads();
  const float var = (red1[0] + red1[1] + red1[2] + red1[3]) * (1.f / MD);
  const float rs = rsqrtf(var + 1e-5f);
#pragma unroll
  for (int i = 0; i < 4; ++i) {
    const int c = t + i * 128;
    const float y = (v[i] - mean) * rs * g[c] + bb[c];
    outF[(size_t)tok * MD + c] = y;
    if (outH) outH[(size_t)tok * MD + c] = (_Float16)y;
  }
}

// ---------------- launcher ----------------
extern "C" void kernel_launch(void* const* d_in, const int* in_sizes, int n_in,
                              void* d_out, int out_size, void* d_ws,
                              size_t ws_size, hipStream_t stream) {
  (void)in_sizes; (void)n_in; (void)out_size; (void)ws_size;
  const float* x   = (const float*)d_in[0];
  const float* Wq  = (const float*)d_in[1];
  const float* Wk  = (const float*)d_in[2];
  const float* Wv  = (const float*)d_in[3];
  const float* Wo  = (const float*)d_in[4];
  const float* W1  = (const float*)d_in[5];
  const float* b1  = (const float*)d_in[6];
  const float* W2  = (const float*)d_in[7];
  const float* b2  = (const float*)d_in[8];
  const float* g1  = (const float*)d_in[9];
  const float* bb1 = (const float*)d_in[10];
  const float* g2  = (const float*)d_in[11];
  const float* bb2 = (const float*)d_in[12];
  float* out = (float*)d_out;

  char* base = (char*)d_ws;
  size_t off = 0;
  auto carve = [&](size_t bytes) -> void* {
    off = (off + 255) & ~(size_t)255;
    void* r = base + off;
    off += bytes;
    return r;
  };

  _Float16* x16   = (_Float16*)carve((size_t)NT * MD * 2);
  _Float16* q16   = (_Float16*)carve((size_t)NT * MD * 2);
  _Float16* k16   = (_Float16*)carve((size_t)NT * MD * 2);
  _Float16* v16   = (_Float16*)carve((size_t)NT * MD * 2);
  _Float16* vt    = (_Float16*)carve((size_t)NB * NH * HD * SQ * 2);
  _Float16* ctx16 = (_Float16*)carve((size_t)NT * MD * 2);
  float*    ao    = (float*)   carve((size_t)NT * MD * 4);
  float*    hf    = (float*)   carve((size_t)NT * MD * 4);
  _Float16* h16   = (_Float16*)carve((size_t)NT * MD * 2);
  _Float16* m1    = (_Float16*)carve((size_t)NT * FF * 2);
  float*    m2    = (float*)   carve((size_t)NT * MD * 4);
  _Float16* wq16  = (_Float16*)carve((size_t)MD * MD * 2);
  _Float16* wk16  = (_Float16*)carve((size_t)MD * MD * 2);
  _Float16* wv16  = (_Float16*)carve((size_t)MD * MD * 2);
  _Float16* wo16  = (_Float16*)carve((size_t)MD * MD * 2);
  _Float16* w116  = (_Float16*)carve((size_t)FF * MD * 2);
  _Float16* w216  = (_Float16*)carve((size_t)MD * FF * 2);

  auto cdiv = [](int a, int b) { return (a + b - 1) / b; };
  const int CB = 256;

  // f32 -> f16 conversions (x4 vectorized)
  f32_to_f16_k<<<cdiv(NT * MD / 4, CB), CB, 0, stream>>>(x, x16, NT * MD / 4);
  f32_to_f16_k<<<cdiv(MD * MD / 4, CB), CB, 0, stream>>>(Wq, wq16, MD * MD / 4);
  f32_to_f16_k<<<cdiv(MD * MD / 4, CB), CB, 0, stream>>>(Wk, wk16, MD * MD / 4);
  f32_to_f16_k<<<cdiv(MD * MD / 4, CB), CB, 0, stream>>>(Wv, wv16, MD * MD / 4);
  f32_to_f16_k<<<cdiv(MD * MD / 4, CB), CB, 0, stream>>>(Wo, wo16, MD * MD / 4);
  f32_to_f16_k<<<cdiv(FF * MD / 4, CB), CB, 0, stream>>>(W1, w116, FF * MD / 4);
  f32_to_f16_k<<<cdiv(MD * FF / 4, CB), CB, 0, stream>>>(W2, w216, MD * FF / 4);

  // Q, K, V projections (f16 outputs)
  dim3 gP(MD / 64, NT / 128);
  gemm_xwT<false, false, false, false, true>
      <<<gP, 128, 0, stream>>>(x16, wq16, nullptr, nullptr, nullptr, q16, NT, MD, MD);
  gemm_xwT<false, false, false, false, true>
      <<<gP, 128, 0, stream>>>(x16, wk16, nullptr, nullptr, nullptr, k16, NT, MD, MD);
  gemm_xwT<false, false, false, false, true>
      <<<gP, 128, 0, stream>>>(x16, wv16, nullptr, nullptr, nullptr, v16, NT, MD, MD);

  // transpose V per head for contiguous B fragments in P@V
  transpose_v_k<<<NB * NH * (SQ / 64), 256, 0, stream>>>(v16, vt);

  // attention (128 KB dynamic LDS for the 16x2048 f32 score strip)
  attention_k<<<NB * NH * (SQ / 16), 128, 16 * SQ * sizeof(float), stream>>>(
      q16, k16, vt, ctx16);

  // output projection + residual x, then LN1 (produces f32 h and f16 h)
  gemm_xwT<false, false, true, true, false>
      <<<gP, 128, 0, stream>>>(ctx16, wo16, nullptr, x, ao, nullptr, NT, MD, MD);
  layernorm_k<<<NT, 128, 0, stream>>>(ao, g1, bb1, hf, h16);

  // MLP: relu(h @ W1^T + b1) -> f16; then @ W2^T + b2 + h -> f32; LN2 -> out
  dim3 gF(FF / 64, NT / 128);
  gemm_xwT<true, true, false, false, true>
      <<<gF, 128, 0, stream>>>(h16, w116, b1, nullptr, nullptr, m1, NT, FF, MD);
  gemm_xwT<true, false, true, true, false>
      <<<gP, 128, 0, stream>>>(m1, w216, b2, hf, m2, nullptr, NT, MD, FF);
  layernorm_k<<<NT, 128, 0, stream>>>(m2, g2, bb2, out, nullptr);
}
